// MaskedFunction_73160472920527
// MI455X (gfx1250) — matrix operand order repaired and checked by reading, not compiled, not run on hardware
//
#include <hip/hip_runtime.h>
#include <stddef.h>
#include <stdint.h>

// ---------------------------------------------------------------------------
// Masked FFN: out = mask ? (gelu(x@W1+b1) @ W2 + b2) : 0
// N=16384 rows, D=1024, F=4096.  bf16 WMMA (v_wmma_f32_16x16x32_bf16),
// active-row compaction (~50% mask => ~2x compute skipped), async LDS staging.
// ---------------------------------------------------------------------------

#define DIM_D 1024
#define DIM_F 4096
#define TM    16      // rows per block tile

typedef __attribute__((ext_vector_type(4)))  __bf16 v4bf;
typedef __attribute__((ext_vector_type(8)))  __bf16 v8bf;
typedef __attribute__((ext_vector_type(16))) __bf16 v16bf;
typedef __attribute__((ext_vector_type(8)))  float  v8f;

union BF16x16 { v16bf v; v8bf h[2]; };

// 16-bit A-fragment (16x32 MxK) from a row-major bf16 row in LDS.
// Lane L (0..15): M=L, K = k0+0..15; lane L+16: K = k0+16..31
// (contiguous K order, matching the B fragments).
__device__ __forceinline__ v16bf load_a_frag(const __bf16* rowbase_k0, int hi) {
  BF16x16 u;
  u.h[0] = *(const v8bf*)(rowbase_k0 + hi * 16);
  u.h[1] = *(const v8bf*)(rowbase_k0 + hi * 16 + 8);
  return u.v;
}

// Branch-free tanh-GELU: tanh(u) = 1 - 2/(1 + exp2(u * 2*log2(e))).
// Saturates correctly for |u| large (exp2 -> inf => rcp -> 0 => tanh -> 1).
__device__ __forceinline__ float fast_gelu(float v) {
  float u = 0.7978845608f * (v + 0.044715f * v * v * v);
  float e = __builtin_amdgcn_exp2f(2.8853900818f * u);   // exp(2u)
  float t = 1.0f - 2.0f * __builtin_amdgcn_rcpf(1.0f + e);
  return 0.5f * v * (1.0f + t);
}

// ---------------------------------------------------------------------------
// Kernel 1: zero the output (masked rows must be exactly 0) + zero counter.
// ---------------------------------------------------------------------------
__global__ void zero_out_kernel(float* __restrict__ out, int n4,
                                int* __restrict__ counter) {
  int i = blockIdx.x * blockDim.x + threadIdx.x;
  if (i == 0) *counter = 0;
  if (i < n4) {
    float4 z = make_float4(0.f, 0.f, 0.f, 0.f);
    ((float4*)out)[i] = z;
  }
}

// ---------------------------------------------------------------------------
// Kernel 2: compact active row indices (scatter-back later is per-row, so the
// atomic ordering nondeterminism does not affect the final output).
// ---------------------------------------------------------------------------
__global__ void compact_mask_kernel(const unsigned char* __restrict__ mask,
                                    int* __restrict__ counter,
                                    int* __restrict__ rowlist, int n) {
  int i = blockIdx.x * blockDim.x + threadIdx.x;
  if (i < n && mask[i] != 0) {
    int pos = atomicAdd(counter, 1);
    rowlist[pos] = i;
  }
}

// ---------------------------------------------------------------------------
// Kernel 3: LDS-tiled transpose + fp32->bf16.  src: SR x SC fp32 row-major,
// dst: SC x SR bf16 row-major.  32x32 tiles, padded to kill bank conflicts;
// coalesced on both the load and the store side.
// ---------------------------------------------------------------------------
__global__ __launch_bounds__(256) void transpose_bf16_kernel(
    const float* __restrict__ src, __bf16* __restrict__ dst, int SR, int SC) {
  __shared__ float tile[32][33];
  int tx = threadIdx.x & 31;
  int ty = threadIdx.x >> 5;            // 0..7
  int r0 = blockIdx.y * 32;             // src row block
  int c0 = blockIdx.x * 32;             // src col block
#pragma unroll
  for (int j = 0; j < 4; ++j)
    tile[ty + 8 * j][tx] = src[(size_t)(r0 + ty + 8 * j) * SC + c0 + tx];
  __syncthreads();
#pragma unroll
  for (int j = 0; j < 4; ++j)
    dst[(size_t)(c0 + ty + 8 * j) * SR + r0 + tx] =
        (__bf16)tile[tx][ty + 8 * j];
}

// ---------------------------------------------------------------------------
// Kernel 4: fused FFN on 16 active rows per block.  8 waves (wave32).
//   stage x (async global->LDS, fp32) -> convert bf16 ->
//   GEMM1 (16x1024 @ 1024x4096) + GELU -> hs (bf16 LDS) ->
//   GEMM2 (16x4096 @ 4096x1024) + b2 -> scatter f32 rows to out.
// Two N-tiles share each A-fragment (2 WMMAs per ds-load pair).
// ---------------------------------------------------------------------------
__global__ __launch_bounds__(256) void ffn_tile16_kernel(
    const float* __restrict__ x, const float* __restrict__ b1,
    const float* __restrict__ b2, const __bf16* __restrict__ w1t,
    const __bf16* __restrict__ w2t, const int* __restrict__ rowlist,
    const int* __restrict__ counter, float* __restrict__ out) {
  __shared__ __align__(16) __bf16 hs[TM * DIM_F];  // 128 KB (h matrix)
  __shared__ __align__(16) __bf16 xs[TM * DIM_D];  // 32 KB  (x tile, bf16)
  __shared__ int rows[TM];
  // fp32 staging buffer aliases hs[0 .. 64KB): consumed before GEMM1 writes hs
  float* xstage = (float*)hs;

  const int tid = threadIdx.x;

  if (tid < TM) {
    int g = blockIdx.x * TM + tid;
    int cnt = *counter;
    rows[tid] = (g < cnt) ? rowlist[g] : -1;
  }
  __syncthreads();
  if (rows[0] < 0) return;  // tile fully beyond active count (uniform)

  // --- async-stage x tile (fp32) into LDS: 16 threads/row, b128 chunks ---
  {
    int r  = tid >> 4;
    int cs = (tid & 15) * 64;
    int row = rows[r];
    const float* src = x + (size_t)((row >= 0) ? row : 0) * DIM_D;
#pragma unroll
    for (int i = 0; i < 16; ++i) {
      int c = cs + i * 4;
      unsigned int       ldsoff = (unsigned int)(size_t)(xstage + r * DIM_D + c);
      unsigned long long gaddr  = (unsigned long long)(size_t)(src + c);
      asm volatile("global_load_async_to_lds_b128 %0, %1, off"
                   :: "v"(ldsoff), "v"(gaddr) : "memory");
    }
    asm volatile("s_wait_asynccnt 0x0" ::: "memory");
  }
  __syncthreads();

  // --- convert staged fp32 -> bf16 xs ---
  {
    int r  = tid >> 4;
    int cs = (tid & 15) * 64;
#pragma unroll
    for (int i = 0; i < 16; ++i) {
      int c = cs + i * 4;
      float4 v = *(const float4*)(xstage + r * DIM_D + c);
      v4bf p = { (__bf16)v.x, (__bf16)v.y, (__bf16)v.z, (__bf16)v.w };
      *(v4bf*)(xs + r * DIM_D + c) = p;
    }
  }
  __syncthreads();

  const int wave = tid >> 5;   // 0..7
  const int lane = tid & 31;
  const int lo16 = lane & 15;  // N within 16x16 tile
  const int hi   = lane >> 4;  // K-half selector

  // ---------------- GEMM1 + GELU:  h = gelu(x @ W1 + b1) -------------------
  // wave owns F columns [wave*512, wave*512+512): 16 dual-tile iterations.
#pragma unroll 1
  for (int ft = 0; ft < 32; ft += 2) {
    int f0 = wave * 512 + ft * 16;
    const __bf16* bc0 = w1t + (size_t)(f0 + lo16) * DIM_D + hi * 16;
    const __bf16* bc1 = bc0 + 16 * DIM_D;
    if (ft < 30)  // prefetch the panel after next (global_prefetch_b8)
      __builtin_prefetch(w1t + (size_t)(f0 + 32 + lo16) * DIM_D, 0, 1);
    v8f acc0 = {}, acc1 = {};
    const __bf16* arow = xs + lo16 * DIM_D;
#pragma unroll 2
    for (int k0 = 0; k0 < DIM_D; k0 += 32) {
      v16bf a  = load_a_frag(arow + k0, hi);
      v16bf b0 = *(const v16bf*)(bc0 + k0);   // 32 contiguous bytes per lane
      v16bf b1v = *(const v16bf*)(bc1 + k0);
      acc0 = __builtin_amdgcn_wmma_f32_16x16x32_bf16(
          false, a, false, b0, (short)0, acc0, false, false);
      acc1 = __builtin_amdgcn_wmma_f32_16x16x32_bf16(
          false, a, false, b1v, (short)0, acc1, false, false);
    }
    float bias0 = b1[f0 + lo16];
    float bias1 = b1[f0 + 16 + lo16];
#pragma unroll
    for (int r = 0; r < 8; ++r) {
      int m = r + 8 * hi;                       // C/D layout: M = r + 8*hi
      hs[m * DIM_F + f0 + lo16]      = (__bf16)fast_gelu(acc0[r] + bias0);
      hs[m * DIM_F + f0 + 16 + lo16] = (__bf16)fast_gelu(acc1[r] + bias1);
    }
  }
  __syncthreads();

  // ---------------- GEMM2:  y = h @ W2 + b2, scatter to out ----------------
  // wave owns D columns [wave*128, wave*128+128): 4 dual-tile iterations.
#pragma unroll 1
  for (int dt = 0; dt < 8; dt += 2) {
    int d0 = wave * 128 + dt * 16;
    const __bf16* bc0 = w2t + (size_t)(d0 + lo16) * DIM_F + hi * 16;
    const __bf16* bc1 = bc0 + 16 * DIM_F;
    if (dt < 6)
      __builtin_prefetch(w2t + (size_t)(d0 + 32 + lo16) * DIM_F, 0, 1);
    v8f acc0 = {}, acc1 = {};
    const __bf16* arow = hs + lo16 * DIM_F;
#pragma unroll 2
    for (int k0 = 0; k0 < DIM_F; k0 += 32) {
      v16bf a  = load_a_frag(arow + k0, hi);
      v16bf b0 = *(const v16bf*)(bc0 + k0);
      v16bf b1v = *(const v16bf*)(bc1 + k0);
      acc0 = __builtin_amdgcn_wmma_f32_16x16x32_bf16(
          false, a, false, b0, (short)0, acc0, false, false);
      acc1 = __builtin_amdgcn_wmma_f32_16x16x32_bf16(
          false, a, false, b1v, (short)0, acc1, false, false);
    }
    float bias0 = b2[d0 + lo16];
    float bias1 = b2[d0 + 16 + lo16];
#pragma unroll
    for (int r = 0; r < 8; ++r) {
      int m = r + 8 * hi;
      int row = rows[m];
      if (row >= 0) {
        out[(size_t)row * DIM_D + d0 + lo16]      = acc0[r] + bias0;
        out[(size_t)row * DIM_D + d0 + 16 + lo16] = acc1[r] + bias1;
      }
    }
  }
}

// ---------------------------------------------------------------------------
extern "C" void kernel_launch(void* const* d_in, const int* in_sizes, int n_in,
                              void* d_out, int out_size, void* d_ws,
                              size_t ws_size, hipStream_t stream) {
  const float*         x    = (const float*)d_in[0];
  const unsigned char* mask = (const unsigned char*)d_in[1];  // jax bool = 1B
  const float*         W1   = (const float*)d_in[2];
  const float*         b1   = (const float*)d_in[3];
  const float*         W2   = (const float*)d_in[4];
  const float*         b2   = (const float*)d_in[5];
  float*               out  = (float*)d_out;

  const int N = in_sizes[1];  // B*T = 16384 rows

  // workspace: [W1T bf16 (F*D)] [W2T bf16 (D*F)] [counter pad] [rowlist]
  char* ws = (char*)d_ws;
  __bf16* w1t = (__bf16*)ws;
  __bf16* w2t = (__bf16*)(ws + (size_t)DIM_F * DIM_D * sizeof(__bf16));
  int* counter = (int*)(ws + 2 * (size_t)DIM_F * DIM_D * sizeof(__bf16));
  int* rowlist = counter + 4;

  // 1) zero output + counter
  {
    int n4 = out_size / 4;
    zero_out_kernel<<<(n4 + 255) / 256, 256, 0, stream>>>(out, n4, counter);
  }
  // 2) compact active rows
  compact_mask_kernel<<<(N + 255) / 256, 256, 0, stream>>>(mask, counter,
                                                           rowlist, N);
  // 3) W1 (DxF) -> W1T (FxD bf16);  4) W2 (FxD) -> W2T (DxF bf16)
  {
    dim3 g1(DIM_F / 32, DIM_D / 32);
    transpose_bf16_kernel<<<g1, 256, 0, stream>>>(W1, w1t, DIM_D, DIM_F);
    dim3 g2(DIM_D / 32, DIM_F / 32);
    transpose_bf16_kernel<<<g2, 256, 0, stream>>>(W2, w2t, DIM_F, DIM_D);
  }
  // 5) fused FFN over worst-case row tiles; inactive tiles exit early
  ffn_tile16_kernel<<<N / TM, 256, 0, stream>>>(x, b1, b2, w1t, w2t, rowlist,
                                                counter, out);
}
